// Clements_69088843923812
// MI455X (gfx1250) — matrix operand order repaired and checked by reading, not compiled, not run on hardware
//
#include <hip/hip_runtime.h>
#include <hip/hip_bf16.h>

#define DIM       256
#define HALFDIM   128
#define NLAYERS    64
#define NROWS   32768

typedef __attribute__((ext_vector_type(16))) _Float16 v16h;
typedef __attribute__((ext_vector_type(8)))  float    v8f;
typedef __attribute__((ext_vector_type(4)))  float    v4f;

union U16H { v16h v; _Float16 h[16]; };
union UF16 { v4f v[4]; float f[16]; };

// ---------------------------------------------------------------------------
// Kernel 1: compose the 64 rotation layers into a single 256x256 matrix M,
// defined by out_row = x_row @ M.  Row r of M = e_r pushed through all layers.
// One block per row; the row lives in LDS; pairs are disjoint per layer.
// ---------------------------------------------------------------------------
__global__ __launch_bounds__(HALFDIM) void compose_kernel(
    const float* __restrict__ theta, float* __restrict__ M) {
  __shared__ float row[DIM];
  const int r = blockIdx.x;
  const int t = threadIdx.x;                       // 0..127
  row[2 * t]     = (2 * t     == r) ? 1.0f : 0.0f;
  row[2 * t + 1] = (2 * t + 1 == r) ? 1.0f : 0.0f;
  __syncthreads();
  for (int l = 0; l < NLAYERS; ++l) {
    if ((l & 1) == 0) {
      // pairs (2t, 2t+1), npairs = 128
      float th = theta[l * HALFDIM + t];
      float s, c;
      sincosf(2.0f * th, &s, &c);
      float xi = row[2 * t], xj = row[2 * t + 1];
      row[2 * t]     = c * xi + s * xj;
      row[2 * t + 1] = s * xi - c * xj;
    } else if (t < HALFDIM - 1) {
      // pairs (2t+1, 2t+2), npairs = 127, endpoints untouched
      float th = theta[l * HALFDIM + t];
      float s, c;
      sincosf(2.0f * th, &s, &c);
      float xi = row[2 * t + 1], xj = row[2 * t + 2];
      row[2 * t + 1] = c * xi + s * xj;
      row[2 * t + 2] = s * xi - c * xj;
    }
    __syncthreads();
  }
  M[r * DIM + 2 * t]     = row[2 * t];
  M[r * DIM + 2 * t + 1] = row[2 * t + 1];
}

// ---------------------------------------------------------------------------
// Kernel 2: split M into f16 hi/lo and pre-swizzle into the CDNA5 WMMA
// B-operand layout for 16-bit 32x16 tiles:
//   lane l: N = ntile*16 + (l&15), K = ktile*32 + (l>=16 ? 8 : 0) + e + (e>=8 ? 8 : 0)
//   element e (0..15) packs VGPR e/2 half e%2 -> contiguous 16 halves / lane.
// Memory layout: [ktile(8)][ntile(16)][lane(32)][elem(16)] f16 (32B per lane).
// ---------------------------------------------------------------------------
__global__ __launch_bounds__(128) void split_kernel(
    const float* __restrict__ M,
    _Float16* __restrict__ Bhi, _Float16* __restrict__ Blo) {
  const int tid   = blockIdx.x * blockDim.x + threadIdx.x;  // 0..4095
  const int tile  = tid >> 5;                               // ktile*16 + ntile
  const int lane  = tid & 31;
  const int ktile = tile >> 4;
  const int ntile = tile & 15;
  const int n     = ntile * 16 + (lane & 15);
  const int kbase = ktile * 32 + ((lane >> 4) << 3);
  const size_t off = ((size_t)tile * 32 + lane) * 16;
#pragma unroll
  for (int e = 0; e < 16; ++e) {
    float v      = M[(size_t)(kbase + e + (e >= 8 ? 8 : 0)) * DIM + n];
    _Float16 h   = (_Float16)v;
    Bhi[off + e] = h;
    Blo[off + e] = (_Float16)(v - (float)h);
  }
}

// ---------------------------------------------------------------------------
// Kernel 3: out = x @ M via split-precision f16 WMMA (3 products, lo*lo
// dropped).  8 waves / block; each wave computes a 32x64 tile (2 M-tiles x
// 4 N-tiles).  Block tile = 64 rows x 256 cols -> 512 blocks.
// A layout per lane (16-bit 16x32): elems 0..7 = K(laneHi*8 + 0..7),
// elems 8..15 = K(16 + laneHi*8 + 0..7), row = lane&15.
// ---------------------------------------------------------------------------
__global__ __launch_bounds__(256) void gemm_kernel(
    const float* __restrict__ x,
    const _Float16* __restrict__ Bhi, const _Float16* __restrict__ Blo,
    float* __restrict__ out) {
  const int lane    = threadIdx.x & 31;
  const int wave    = threadIdx.x >> 5;   // 0..7
  const int mgroup  = wave >> 2;          // 0..1 -> 32-row half
  const int ngroup  = wave & 3;           // 0..3 -> 64-col quarter
  const int lane16  = lane & 15;
  const int laneHi  = lane >> 4;
  const int rowBase = blockIdx.x * 64 + mgroup * 32;
  const int ntg0    = ngroup * 4;         // first global N-tile index

  v8f acc[2][4] = {};

  for (int kk = 0; kk < 8; ++kk) {
    const int k0 = kk * 32 + laneHi * 8;

    // ---- load x tiles (A operand) and split into f16 hi/lo ----
    v16h ahi[2], alo[2];
#pragma unroll
    for (int mt = 0; mt < 2; ++mt) {
      const float* xp = x + (size_t)(rowBase + mt * 16 + lane16) * DIM + k0;
      if (kk < 7) __builtin_prefetch(xp + 32, 0, 0);  // next K chunk
      UF16 uf;
      uf.v[0] = *(const v4f*)(xp + 0);
      uf.v[1] = *(const v4f*)(xp + 4);
      uf.v[2] = *(const v4f*)(xp + 16);
      uf.v[3] = *(const v4f*)(xp + 20);
      U16H ah, al;
#pragma unroll
      for (int e = 0; e < 16; ++e) {
        float    fv = uf.f[e];
        _Float16 h  = (_Float16)fv;
        ah.h[e] = h;
        al.h[e] = (_Float16)(fv - (float)h);
      }
      ahi[mt] = ah.v;
      alo[mt] = al.v;
    }

    // ---- B tiles (pre-swizzled) + WMMA ----
#pragma unroll
    for (int nt = 0; nt < 4; ++nt) {
      const size_t boff = (((size_t)kk * 16 + (ntg0 + nt)) * 32 + lane) * 16;
      v16h bh = *(const v16h*)(Bhi + boff);
      v16h bl = *(const v16h*)(Blo + boff);
#pragma unroll
      for (int mt = 0; mt < 2; ++mt) {
        acc[mt][nt] = __builtin_amdgcn_wmma_f32_16x16x32_f16(
            false, ahi[mt], false, bh, (short)0, acc[mt][nt], false, false);
        acc[mt][nt] = __builtin_amdgcn_wmma_f32_16x16x32_f16(
            false, alo[mt], false, bh, (short)0, acc[mt][nt], false, false);
        acc[mt][nt] = __builtin_amdgcn_wmma_f32_16x16x32_f16(
            false, ahi[mt], false, bl, (short)0, acc[mt][nt], false, false);
      }
    }
  }

  // ---- store C: VGPR j -> row j + 8*laneHi, col = lane16 ----
#pragma unroll
  for (int mt = 0; mt < 2; ++mt) {
#pragma unroll
    for (int nt = 0; nt < 4; ++nt) {
#pragma unroll
      for (int j = 0; j < 8; ++j) {
        const int r = rowBase + mt * 16 + laneHi * 8 + j;
        const int c = ngroup * 64 + nt * 16 + lane16;
        out[(size_t)r * DIM + c] = acc[mt][nt][j];
      }
    }
  }
}

// ---------------------------------------------------------------------------
extern "C" void kernel_launch(void* const* d_in, const int* in_sizes, int n_in,
                              void* d_out, int out_size, void* d_ws, size_t ws_size,
                              hipStream_t stream) {
  const float* x     = (const float*)d_in[0];
  const float* theta = (const float*)d_in[1];
  float*       out   = (float*)d_out;

  char*     ws  = (char*)d_ws;
  float*    M   = (float*)ws;                                  // 256 KB
  _Float16* Bhi = (_Float16*)(ws + (size_t)DIM * DIM * 4);     // 128 KB
  _Float16* Blo = Bhi + (size_t)DIM * DIM;                     // 128 KB

  compose_kernel<<<DIM, HALFDIM, 0, stream>>>(theta, M);
  split_kernel<<<32, 128, 0, stream>>>(M, Bhi, Blo);
  gemm_kernel<<<NROWS / 64, 256, 0, stream>>>(x, Bhi, Blo, out);
}
